// RNN_57930518889094
// MI455X (gfx1250) — matrix operand order repaired and checked by reading, not compiled, not run on hardware
//
#include <hip/hip_runtime.h>

// Problem constants (from reference)
#define T_ 1024
#define B_ 256
#define I_ 128
#define H_ 256

// Padded LDS row strides (in ushorts) to avoid bank conflicts on the
// b128 A-fragment loads: stride bytes % 256 must not be 0.
// 528 B / 272 B -> lane ln hits banks 4*ln..4*ln+3 (conflict-free).
#define HSTRIDE 264   // for 16 x H   h buffer
#define XSTRIDE 136   // for 16 x I   x buffer

typedef __attribute__((ext_vector_type(16))) __bf16 v16bf;
typedef __attribute__((ext_vector_type(8)))  float  v8f;

// Union to assemble WMMA bf16 fragments from raw 128-bit loads.
union V16 {
    v16bf v;
    unsigned short s[16];
    uint4 q[2];
};

// float -> bf16 bits via native RNE fptrunc (lowers to v_cvt_pk_bf16_f32).
__device__ __forceinline__ unsigned short f2bf(float f) {
    __bf16 b = (__bf16)f;
    return __builtin_bit_cast(unsigned short, b);
}

// Branchless tanh: 1 - 2/(e^{2x}+1); exact +-1 saturation, NaN-free.
__device__ __forceinline__ float fast_tanh(float x) {
    float e = __expf(2.0f * x);
    return 1.0f - 2.0f * __builtin_amdgcn_rcpf(e + 1.0f);
}

// ---------------------------------------------------------------------------
// Prep: convert Wi [H,I] and Wh [H,H] f32 -> bf16 into d_ws (once per call).
// ws layout: [0, H*I) = Wi_bf ; [H*I, H*I + H*H) = Wh_bf (ushort units).
// ---------------------------------------------------------------------------
__global__ __launch_bounds__(256)
void cvt_weights_kernel(const float* __restrict__ Wi,
                        const float* __restrict__ Wh,
                        unsigned short* __restrict__ ws) {
    const int NWi = H_ * I_;              // 32768
    const int idx = (blockIdx.x * 256 + threadIdx.x) * 4;
    float4 v;
    if (idx < NWi) v = *(const float4*)(Wi + idx);
    else           v = *(const float4*)(Wh + (idx - NWi));
    ushort4 o;
    o.x = f2bf(v.x); o.y = f2bf(v.y); o.z = f2bf(v.z); o.w = f2bf(v.w);
    *(ushort4*)(ws + idx) = o;
}

// ---------------------------------------------------------------------------
// Phase 1: xproj[t,b,n] = sum_k X[b,t,k]*Wi[n,k] + bh[n] -> d_out [T,B,H].
// Block = 512 threads = 16 waves. X tile staged ONCE per block into LDS
// as bf16; each wave owns one 16x16 N tile, K-loop = 4 WMMAs.
// Bias is folded into the accumulator init.
// ---------------------------------------------------------------------------
__global__ __launch_bounds__(512)
void xproj_kernel(const float* __restrict__ X,
                  const unsigned short* __restrict__ Wi_bf,
                  const float* __restrict__ bh,
                  float* __restrict__ out) {
    __shared__ unsigned short xbuf[16 * XSTRIDE];   // 16 x 128 bf16, padded

    const int lane = threadIdx.x & 31;
    const int wave = threadIdx.x >> 5;   // 0..15 -> N tile
    const int n0   = wave << 4;
    const int m0   = blockIdx.x << 4;    // row tile in flattened (b*T + t)
    const int b    = m0 >> 10;           // / T
    const int t0   = m0 & (T_ - 1);
    const int ln   = lane & 15;
    const int hi   = lane >> 4;          // half-wave select

    // Cooperative stage: 2048 f32 -> bf16, 4 elements per thread.
    {
        const int idx = threadIdx.x * 4;          // 0..2044
        const int row = idx >> 7;
        const int col = idx & (I_ - 1);
        float4 v = *(const float4*)(X + (size_t)(m0 + row) * I_ + col);
        ushort4 o;
        o.x = f2bf(v.x); o.y = f2bf(v.y); o.z = f2bf(v.z); o.w = f2bf(v.w);
        *(ushort4*)(xbuf + row * XSTRIDE + col) = o;
    }

    // B fragments (Wi bf16): lane holds n = n0+ln, k = hi*8+{0..7} and 16+hi*8+{0..7}.
    V16 Bt[4];
    {
        const unsigned short* wrow = Wi_bf + (n0 + ln) * I_;
#pragma unroll
        for (int kc = 0; kc < 4; ++kc) {
            const unsigned short* p = wrow + kc * 32 + hi * 8;
            Bt[kc].q[0] = *(const uint4*)p;
            Bt[kc].q[1] = *(const uint4*)(p + 16);
        }
    }

    // Accumulator starts as broadcast bias (C-fragment: col = ln for all i).
    const float bias = bh[n0 + ln];
    v8f c;
#pragma unroll
    for (int i = 0; i < 8; ++i) c[i] = bias;

    __syncthreads();

    // A fragments from LDS; 4 chained WMMAs over K=128.
#pragma unroll
    for (int kc = 0; kc < 4; ++kc) {
        V16 a;
        const unsigned short* p = xbuf + ln * XSTRIDE + kc * 32 + hi * 8;
        a.q[0] = *(const uint4*)p;          // ds_load_b128
        a.q[1] = *(const uint4*)(p + 16);   // ds_load_b128
        c = __builtin_amdgcn_wmma_f32_16x16x32_bf16(
                false, a.v, false, Bt[kc].v, (short)0, c, false, false);
    }

    // C tile: VGPR i -> row i + hi*8, col ln. Store to out[T,B,H].
#pragma unroll
    for (int i = 0; i < 8; ++i) {
        const int row = i + hi * 8;               // local t offset
        out[(size_t)(t0 + row) * (B_ * H_) + b * H_ + n0 + ln] = c[i];
    }
}

// ---------------------------------------------------------------------------
// Phase 2: sequential scan. One block per 16-row batch tile (16 blocks),
// 512 threads = 16 waves, wave w owns N tile [16w, 16w+16).
// Wh bf16 B-fragments register-resident for the whole T loop.
// h is double-buffered in LDS (one barrier per step); the xproj tile is
// the C-operand init of chain 0; two independent 4-deep WMMA chains.
// ---------------------------------------------------------------------------
__global__ __launch_bounds__(512)
void rnn_scan_kernel(const unsigned short* __restrict__ Wh_bf,
                     float* __restrict__ out) {
    __shared__ unsigned short hbuf[2][16 * HSTRIDE];

    const int lane = threadIdx.x & 31;
    const int wave = threadIdx.x >> 5;
    const int n0   = wave << 4;
    const int ln   = lane & 15;
    const int hi   = lane >> 4;
    const int b0   = blockIdx.x << 4;

    // Preload Wh B fragments (bf16, pre-converted): 8 K-chunks of 32.
    V16 Bt[8];
    {
        const unsigned short* wrow = Wh_bf + (n0 + ln) * H_;
#pragma unroll
        for (int kc = 0; kc < 8; ++kc) {
            const unsigned short* p = wrow + kc * 32 + hi * 8;
            Bt[kc].q[0] = *(const uint4*)p;
            Bt[kc].q[1] = *(const uint4*)(p + 16);
        }
    }

    // h0 = 0 in buffer 0.
    for (int idx = threadIdx.x; idx < 16 * HSTRIDE; idx += 512)
        hbuf[0][idx] = 0;
    __syncthreads();

    // Per-lane base offset into out[t] for this tile: row i+hi*8, col ln.
    const size_t xbase = (size_t)b0 * H_ + n0 + ln;

    // Preload xproj for t=0.
    float xpv[8];
    {
        const float* p = out + xbase;
#pragma unroll
        for (int i = 0; i < 8; ++i) xpv[i] = p[(i + hi * 8) * H_];
    }

    for (int t = 0; t < T_; ++t) {
        const unsigned short* hcur = hbuf[t & 1];
        unsigned short*       hnxt = hbuf[(t + 1) & 1];

        // Prefetch next timestep's xproj (overlaps the WMMA chains).
        float xpn[8] = {};
        if (t + 1 < T_) {
            const float* p = out + (size_t)(t + 1) * (B_ * H_) + xbase;
#pragma unroll
            for (int i = 0; i < 8; ++i) xpn[i] = p[(i + hi * 8) * H_];
        }

        // Two independent 4-deep WMMA chains; chain 0 starts from xproj tile.
        v8f c0, c1 = {};
#pragma unroll
        for (int i = 0; i < 8; ++i) c0[i] = xpv[i];

#pragma unroll
        for (int kc = 0; kc < 4; ++kc) {
            V16 a0, a1;
            const unsigned short* p = hcur + ln * HSTRIDE + kc * 32 + hi * 8;
            a0.q[0] = *(const uint4*)p;                // ds_load_b128
            a0.q[1] = *(const uint4*)(p + 16);
            a1.q[0] = *(const uint4*)(p + 128);        // kc+4 chunk
            a1.q[1] = *(const uint4*)(p + 144);
            c0 = __builtin_amdgcn_wmma_f32_16x16x32_bf16(
                     false, a0.v, false, Bt[kc].v,     (short)0, c0, false, false);
            c1 = __builtin_amdgcn_wmma_f32_16x16x32_bf16(
                     false, a1.v, false, Bt[kc + 4].v, (short)0, c1, false, false);
        }

        // h_new = tanh(xproj + h@Wh^T); write bf16 h to the other buffer.
        float hnew[8];
#pragma unroll
        for (int i = 0; i < 8; ++i) {
            hnew[i] = fast_tanh(c0[i] + c1[i]);
            hnxt[(i + hi * 8) * HSTRIDE + n0 + ln] = f2bf(hnew[i]);
        }
        __syncthreads();   // h_new visible; prev-buffer reads already done

        // f32 result store (off the critical path: next step needs only LDS).
        float* po = out + (size_t)t * (B_ * H_) + xbase;
#pragma unroll
        for (int i = 0; i < 8; ++i) po[(i + hi * 8) * H_] = hnew[i];

#pragma unroll
        for (int i = 0; i < 8; ++i) xpv[i] = xpn[i];
    }
}

// ---------------------------------------------------------------------------
extern "C" void kernel_launch(void* const* d_in, const int* in_sizes, int n_in,
                              void* d_out, int out_size, void* d_ws, size_t ws_size,
                              hipStream_t stream) {
    const float* X  = (const float*)d_in[0];   // [B, T, I]
    const float* Wi = (const float*)d_in[1];   // [H, I]
    const float* Wh = (const float*)d_in[2];   // [H, H]
    const float* bh = (const float*)d_in[3];   // [H]
    float* out = (float*)d_out;                // [T, B, H]

    unsigned short* wbf   = (unsigned short*)d_ws;
    unsigned short* Wi_bf = wbf;               // H*I ushorts
    unsigned short* Wh_bf = wbf + H_ * I_;     // H*H ushorts

    // Prep: weights f32 -> bf16 (96 blocks x 256 threads x 4 elems = 98304).
    cvt_weights_kernel<<<(H_ * I_ + H_ * H_) / (256 * 4), 256, 0, stream>>>(
        Wi, Wh, wbf);
    // Phase 1: xproj -> d_out (16384 M-tiles, 16 waves each).
    xproj_kernel<<<(B_ * T_) / 16, 512, 0, stream>>>(X, Wi_bf, bh, out);
    // Phase 2: 16 independent batch tiles scan T steps in-place.
    rnn_scan_kernel<<<B_ / 16, 512, 0, stream>>>(Wh_bf, out);
}
